// CellStacker_56186762166640
// MI455X (gfx1250) — matrix operand order repaired and checked by reading, not compiled, not run on hardware
//
#include <hip/hip_runtime.h>
#include <hip/hip_bf16.h>

typedef __attribute__((ext_vector_type(16))) __bf16 v16bf;
typedef __attribute__((ext_vector_type(8)))  float  v8f;
typedef __attribute__((ext_vector_type(4)))  unsigned int u32x4;
typedef __attribute__((ext_vector_type(4)))  float  f32x4;

union FragAB { v16bf v; u32x4 q[2]; };

#define T_STEPS 256
#define BATCH   64
#define DIM     1024   // D == H
#define KDIM    2048   // D + H
#define GATES4  4096   // 4*H
#define NLAYERS 4
#define KSPLIT  512    // K per wave (4-way split of KDIM)

__device__ __forceinline__ float sigf(float x) {
  return 1.0f / (1.0f + __expf(-x));
}

// ---- fp32 -> bf16 weight repack: Wcat[l][row][k] = k<D ? Wih[l][row][k] : Whh[l][row][k-D]
__global__ void repack_weights(const float* __restrict__ Wih,
                               const float* __restrict__ Whh,
                               __bf16* __restrict__ Wcat, long total) {
  long e = (long)blockIdx.x * blockDim.x + threadIdx.x;
  if (e >= total) return;
  int  k  = (int)(e % KDIM);
  long lr = e / KDIM;                     // l*4096 + row
  float v = (k < DIM) ? Wih[lr * DIM + k] : Whh[lr * DIM + (k - DIM)];
  Wcat[e] = (__bf16)v;
}

// ---- one-shot fp32 -> bf16 copy (whole input sequence, and reused for h0 via init)
__global__ void cvt_f32_bf16(const float* __restrict__ src, __bf16* __restrict__ dst, long total) {
  long e = (long)blockIdx.x * blockDim.x + threadIdx.x;
  if (e < total) dst[e] = (__bf16)src[e];
}

// ---- init: c_state = c0 ; Hh slot0 = bf16(h0)
__global__ void init_state(const float* __restrict__ h0, const float* __restrict__ c0,
                           __bf16* __restrict__ Hh0, float* __restrict__ c_state, int total) {
  int e = blockIdx.x * blockDim.x + threadIdx.x;
  if (e >= total) return;
  Hh0[e] = (__bf16)h0[e];
  c_state[e] = c0[e];
}

// ---- fused GEMM (WMMA bf16, f32 acc, K split 4-way across waves) + LSTM cell
// grid = 32 blocks x 1024 threads (32 waves): wave -> m = w&3, jsub = (w>>2)&1, kw = w>>3
// kw 0,1 accumulate the x-half (Xsrc), kw 2,3 the h-half (Hsrc); LDS reduce; kw==0 does cell.
__global__ __launch_bounds__(1024)
void lstm_layer_step(const __bf16* __restrict__ Xsrc,   // [B][DIM] bf16 input rows, this layer
                     const __bf16* __restrict__ Hsrc,   // [B][DIM] bf16 h(t-1), this layer
                     const __bf16* __restrict__ W,      // [4096][KDIM] bf16 weights, this layer
                     const float*  __restrict__ bias,   // [4096]
                     float*  __restrict__ c_state,      // [B][H] this layer
                     __bf16* __restrict__ Hnext,        // h -> other slot, this layer
                     __bf16* __restrict__ Xnext,        // h -> next layer input (or null)
                     float*  __restrict__ out_h,        // out_seq[t] base (layer 3 only)
                     float*  __restrict__ hT_out,       // h_T[l] base (t==T-1 only)
                     float*  __restrict__ cT_out)       // c_T[l] base (t==T-1 only)
{
  extern __shared__ float red[];                 // [8 groups][3 partials][4 gates][256 floats]
  const int lane  = threadIdx.x & 31;
  const int wave  = threadIdx.x >> 5;            // 0..31
  const int m     = wave & 3;                    // batch tile: rows m*16..m*16+15
  const int jsub  = (wave >> 2) & 1;
  const int kw    = wave >> 3;                   // 0..3 : K quarter
  const int g8    = wave & 7;                    // output-tile group (m,jsub)
  const int jbase = blockIdx.x * 32 + jsub * 16;
  const int n     = lane & 15;
  const int col   = jbase + n;

  // A frag (16-bit A 16x32): lane<16 -> K0..15, lane>=16 -> K16..31
  const int aklo = (lane < 16) ? 0 : 16;
  // B frag (K-major cols): lane<16 holds K0..15 of column n, lane>=16 holds K16..31
  const int bklo = (lane < 16) ? 0 : 16;
  const int row  = m * 16 + n;

  const __bf16* abase = ((kw < 2) ? (Xsrc + (size_t)row * DIM + kw * KSPLIT)
                                  : (Hsrc + (size_t)row * DIM + (kw - 2) * KSPLIT)) + aklo;
  const size_t koff = (size_t)kw * KSPLIT + bklo;
  const __bf16* b0r = W + (size_t)(0 * DIM + col) * KDIM + koff;
  const __bf16* b1r = W + (size_t)(1 * DIM + col) * KDIM + koff;
  const __bf16* b2r = W + (size_t)(2 * DIM + col) * KDIM + koff;
  const __bf16* b3r = W + (size_t)(3 * DIM + col) * KDIM + koff;

  v8f acc0 = {}, acc1 = {}, acc2 = {}, acc3 = {};

  for (int kk = 0; kk < KSPLIT; kk += 32) {
    FragAB a, b0, b1, b2, b3;
    a.q[0]  = *(const u32x4*)(abase + kk);
    a.q[1]  = *(const u32x4*)(abase + kk + 8);
    b0.q[0] = *(const u32x4*)(b0r + kk);
    b0.q[1] = *(const u32x4*)(b0r + kk + 8);
    b1.q[0] = *(const u32x4*)(b1r + kk);
    b1.q[1] = *(const u32x4*)(b1r + kk + 8);
    b2.q[0] = *(const u32x4*)(b2r + kk);
    b2.q[1] = *(const u32x4*)(b2r + kk + 8);
    b3.q[0] = *(const u32x4*)(b3r + kk);
    b3.q[1] = *(const u32x4*)(b3r + kk + 8);
    acc0 = __builtin_amdgcn_wmma_f32_16x16x32_bf16(false, a.v, false, b0.v, (short)0, acc0, false, false);
    acc1 = __builtin_amdgcn_wmma_f32_16x16x32_bf16(false, a.v, false, b1.v, (short)0, acc1, false, false);
    acc2 = __builtin_amdgcn_wmma_f32_16x16x32_bf16(false, a.v, false, b2.v, (short)0, acc2, false, false);
    acc3 = __builtin_amdgcn_wmma_f32_16x16x32_bf16(false, a.v, false, b3.v, (short)0, acc3, false, false);
  }

  // K-partial reduction through LDS: waves kw>0 deposit, kw==0 accumulates.
  if (kw > 0) {
    float* base = &red[(size_t)((g8 * 3 + (kw - 1)) * 4) * 256 + lane * 8];
    union { v8f v; f32x4 q[2]; } u;
    u.v = acc0; *(f32x4*)(base + 0 * 256) = u.q[0]; *(f32x4*)(base + 0 * 256 + 4) = u.q[1];
    u.v = acc1; *(f32x4*)(base + 1 * 256) = u.q[0]; *(f32x4*)(base + 1 * 256 + 4) = u.q[1];
    u.v = acc2; *(f32x4*)(base + 2 * 256) = u.q[0]; *(f32x4*)(base + 2 * 256 + 4) = u.q[1];
    u.v = acc3; *(f32x4*)(base + 3 * 256) = u.q[0]; *(f32x4*)(base + 3 * 256 + 4) = u.q[1];
  }
  __syncthreads();
  if (kw > 0) return;

#pragma unroll
  for (int p = 0; p < 3; ++p) {
    const float* base = &red[(size_t)((g8 * 3 + p) * 4) * 256 + lane * 8];
#pragma unroll
    for (int v = 0; v < 8; ++v) {
      acc0[v] += base[0 * 256 + v];
      acc1[v] += base[1 * 256 + v];
      acc2[v] += base[2 * 256 + v];
      acc3[v] += base[3 * 256 + v];
    }
  }

  const float bi = bias[0 * DIM + col];
  const float bf = bias[1 * DIM + col];
  const float bg = bias[2 * DIM + col];
  const float bo = bias[3 * DIM + col];

  // C/D 16x16 f32 layout: VGPR v holds row M = v (lanes 0-15) / v+8 (lanes 16-31)
  const int rbase = m * 16 + ((lane < 16) ? 0 : 8);

#pragma unroll
  for (int v = 0; v < 8; ++v) {
    const int brow = rbase + v;
    const size_t cidx = (size_t)brow * DIM + col;
    const float ig = sigf(acc0[v] + bi);
    const float fg = sigf(acc1[v] + bf);
    const float gg = tanhf(acc2[v] + bg);
    const float og = sigf(acc3[v] + bo);
    const float cn = fg * c_state[cidx] + ig * gg;
    const float hn = og * tanhf(cn);
    c_state[cidx] = cn;
    Hnext[cidx] = (__bf16)hn;
    if (Xnext)  Xnext[cidx]  = (__bf16)hn;
    if (out_h)  out_h[cidx]  = hn;
    if (hT_out) hT_out[cidx] = hn;
    if (cT_out) cT_out[cidx] = cn;
  }
}

extern "C" void kernel_launch(void* const* d_in, const int* in_sizes, int n_in,
                              void* d_out, int out_size, void* d_ws, size_t ws_size,
                              hipStream_t stream) {
  (void)in_sizes; (void)n_in; (void)out_size; (void)ws_size;
  const float* input_seq = (const float*)d_in[0];   // [T][B][D]
  const float* h0        = (const float*)d_in[1];   // [L][B][H]
  const float* c0        = (const float*)d_in[2];   // [L][B][H]
  const float* Wih       = (const float*)d_in[3];   // [L][4H][D]
  const float* Whh       = (const float*)d_in[4];   // [L][4H][H]
  const float* bias      = (const float*)d_in[5];   // [L][4H]
  float* out = (float*)d_out;

  // workspace carve-up (all L2-resident: ~98 MB of 192 MB L2)
  char* ws = (char*)d_ws;
  __bf16* Wcat = (__bf16*)ws;                                         // 64 MB
  size_t off = (size_t)NLAYERS * GATES4 * KDIM * sizeof(__bf16);
  __bf16* Xbf = (__bf16*)(ws + off);                                  // 32 MB: bf16 input seq
  off += (size_t)T_STEPS * BATCH * DIM * sizeof(__bf16);
  __bf16* Hx = (__bf16*)(ws + off);                                   // 512 KB: per-layer x feed
  off += (size_t)NLAYERS * BATCH * DIM * sizeof(__bf16);
  __bf16* Hh = (__bf16*)(ws + off);                                   // 1 MB: double-buffered h
  off += (size_t)2 * NLAYERS * BATCH * DIM * sizeof(__bf16);
  float* cst = (float*)(ws + off);                                    // 1 MB: c state

  const long wtotal = (long)NLAYERS * GATES4 * KDIM;
  repack_weights<<<(int)((wtotal + 255) / 256), 256, 0, stream>>>(Wih, Whh, Wcat, wtotal);

  const long xtotal = (long)T_STEPS * BATCH * DIM;
  cvt_f32_bf16<<<(int)((xtotal + 255) / 256), 256, 0, stream>>>(input_seq, Xbf, xtotal);

  const int stotal = NLAYERS * BATCH * DIM;
  init_state<<<(stotal + 255) / 256, 256, 0, stream>>>(h0, c0, Hh, cst, stotal);

  const size_t lay = (size_t)BATCH * DIM;
  const size_t slot = (size_t)NLAYERS * lay;
  float* out_seq = out;
  float* hT = out + (size_t)T_STEPS * BATCH * DIM;
  float* cT = hT + (size_t)NLAYERS * BATCH * DIM;
  const size_t ldsBytes = 8 * 3 * 4 * 256 * sizeof(float);            // 96 KB of 320 KB/WGP

  for (int t = 0; t < T_STEPS; ++t) {
    const int s  = t & 1;
    const int s2 = (t + 1) & 1;
    for (int l = 0; l < NLAYERS; ++l) {
      const __bf16* Xsrc = (l == 0) ? (Xbf + (size_t)t * lay) : (Hx + (size_t)l * lay);
      lstm_layer_step<<<32, 1024, ldsBytes, stream>>>(
          Xsrc,
          Hh + s * slot + (size_t)l * lay,
          Wcat + (size_t)l * GATES4 * KDIM,
          bias + (size_t)l * GATES4,
          cst + (size_t)l * lay,
          Hh + s2 * slot + (size_t)l * lay,
          (l < NLAYERS - 1) ? (Hx + (size_t)(l + 1) * lay) : nullptr,
          (l == NLAYERS - 1) ? out_seq + (size_t)t * lay : nullptr,
          (t == T_STEPS - 1) ? hT + (size_t)l * lay : nullptr,
          (t == T_STEPS - 1) ? cT + (size_t)l * lay : nullptr);
    }
  }
}
